// EdgeTravelTimeGNN_64476049047624
// MI455X (gfx1250) — matrix operand order repaired and checked by reading, not compiled, not run on hardware
//
#include <hip/hip_runtime.h>
#include <hip/hip_bf16.h>
#include <math.h>

// ---------------- problem constants (match reference) ----------------
#define N_NODES 100000
#define N_EDGES 800000
#define NODE_F  5
#define EDGE_F  12
#define HIDDEN  128
#define N_LAYERS 3
#define LN_EPS  1e-5f

typedef __attribute__((ext_vector_type(16))) __bf16 v16bf;
typedef __attribute__((ext_vector_type(8)))  float  v8f;

union ABFrag { v16bf v; unsigned short us[16]; uint4 q[2]; };

__device__ __forceinline__ unsigned short f2bf(float f) {
  unsigned int u = __builtin_bit_cast(unsigned int, f);
  unsigned int r = u + 0x7FFFu + ((u >> 16) & 1u);   // round-to-nearest-even
  return (unsigned short)(r >> 16);
}

// ---- CDNA5 async Global->LDS copy (ASYNCcnt-tracked), guarded fallback ----
#if defined(__gfx1250__) && \
    __has_builtin(__builtin_amdgcn_global_load_async_to_lds_b128) && \
    __has_builtin(__builtin_amdgcn_s_wait_asynccnt)
#define ASYNC_LDS 1
#else
#define ASYNC_LDS 0
#endif

#if ASYNC_LDS
// builtin prototype (from hipcc diagnostic): (v4i as1*, v4i as3*, imm int, imm int)
typedef int v4i_vs __attribute__((vector_size(16)));
#define AS1_V4I(p) ((__attribute__((address_space(1))) v4i_vs*)(void*)(p))
#define AS3_V4I(p) ((__attribute__((address_space(3))) v4i_vs*)(void*)(p))
#endif

__device__ __forceinline__ void copy16_g2lds(const unsigned short* g, unsigned short* l) {
#if ASYNC_LDS
  __builtin_amdgcn_global_load_async_to_lds_b128(AS1_V4I(g), AS3_V4I(l), 0, 0);
#else
  *(uint4*)l = *(const uint4*)g;
#endif
}

__device__ __forceinline__ void async_fence() {
#if ASYNC_LDS
  __builtin_amdgcn_s_wait_asynccnt(0);
#endif
}

// ---------------- utility kernels ----------------
__global__ void k_zero4(float4* p, int n4) {
  int i = blockIdx.x * blockDim.x + threadIdx.x;
  if (i < n4) p[i] = make_float4(0.f, 0.f, 0.f, 0.f);
}

__global__ void k_degree(const int* __restrict__ ei, float* __restrict__ deg) {
  int e = blockIdx.x * blockDim.x + threadIdx.x;
  if (e < N_EDGES) atomicAdd(&deg[ei[N_EDGES + e]], 1.0f);
}

__global__ void k_invdeg(float* d) {
  int i = blockIdx.x * blockDim.x + threadIdx.x;
  if (i < N_NODES) { float v = d[i]; d[i] = (v > 0.f) ? 1.0f / v : 0.0f; }
}

// h0 = x @ node_W + node_b  (K=5; memory bound). Writes fp32 h + bf16 shadow.
__global__ void k_node_proj(const float* __restrict__ x, const float* __restrict__ W,
                            const float* __restrict__ b, float* __restrict__ h,
                            unsigned short* __restrict__ hbf) {
  int tid = blockIdx.x * blockDim.x + threadIdx.x;
  if (tid >= N_NODES * HIDDEN) return;
  int n = tid >> 7, f = tid & 127;
  const float* xr = x + n * NODE_F;
  float acc = b[f];
#pragma unroll
  for (int k = 0; k < NODE_F; ++k) acc += xr[k] * W[k * HIDDEN + f];
  h[tid] = acc;
  hbf[tid] = f2bf(acc);
}

// gather h[src] rows (fp32), scatter-add into agg[dst] (fp32 atomics)
__global__ void k_scatter(const float* __restrict__ h, const int* __restrict__ ei,
                          float* __restrict__ agg) {
  int t = blockIdx.x * blockDim.x + threadIdx.x;
  if (t >= N_EDGES * 32) return;
  int e = t >> 5, q = t & 31;
  int s = ei[e], d = ei[N_EDGES + e];
  float4 v = *(const float4*)(h + (size_t)s * HIDDEN + q * 4);
  float* a = agg + (size_t)d * HIDDEN + q * 4;
  atomicAdd(a + 0, v.x); atomicAdd(a + 1, v.y);
  atomicAdd(a + 2, v.z); atomicAdd(a + 3, v.w);
}

// ---------------- weight prep: fp32 -> bf16 B-fragments ----------------
// B fragment layout (16x16x32 bf16 WMMA): lane = column (n = nt*16 + lane&15),
// lanes 0-15 hold K = kt*32 + {0..7,16..23} (2 per VGPR), lanes 16-31 hold K+8.
// Stored as [frag][lane][16 bf16] so every lane loads 32 contiguous bytes.
#define SAGE_FRAG_ELEMS (N_LAYERS * 2 * 4 * 8 * 512)  // 98304
#define W1_FRAG_ELEMS   (9 * 8 * 512)                 // 36864
#define W2_FRAG_ELEMS   (4 * 4 * 512)                 // 8192

__global__ void k_prep_weights(const float* __restrict__ Wl, const float* __restrict__ Wr,
                               const float* __restrict__ W1, const float* __restrict__ W2,
                               unsigned short* __restrict__ fsage,
                               unsigned short* __restrict__ f1,
                               unsigned short* __restrict__ f2) {
  int tid = blockIdx.x * blockDim.x + threadIdx.x;
  if (tid < SAGE_FRAG_ELEMS) {
    int within = tid & 511, fid = tid >> 9;
    int lane = within >> 4, j = within & 15;
    int nt = fid & 7, kt = (fid >> 3) & 3, mat = (fid >> 5) & 1, layer = fid >> 6;
    int k = kt * 32 + (lane >> 4) * 8 + (j & 7) + (j >> 3) * 16;
    int n = nt * 16 + (lane & 15);
    const float* W = (mat == 0 ? Wl : Wr) + (size_t)layer * HIDDEN * HIDDEN;
    fsage[tid] = f2bf(W[k * HIDDEN + n]);
  } else if (tid < SAGE_FRAG_ELEMS + W1_FRAG_ELEMS) {
    int t = tid - SAGE_FRAG_ELEMS;
    int within = t & 511, fid = t >> 9;
    int lane = within >> 4, j = within & 15;
    int nt = fid & 7, kt = fid >> 3;
    int k = kt * 32 + (lane >> 4) * 8 + (j & 7) + (j >> 3) * 16;
    int n = nt * 16 + (lane & 15);
    f1[t] = (k < 2 * HIDDEN + EDGE_F) ? f2bf(W1[k * HIDDEN + n]) : (unsigned short)0;
  } else if (tid < SAGE_FRAG_ELEMS + W1_FRAG_ELEMS + W2_FRAG_ELEMS) {
    int t = tid - SAGE_FRAG_ELEMS - W1_FRAG_ELEMS;
    int within = t & 511, fid = t >> 9;
    int lane = within >> 4, j = within & 15;
    int nt = fid & 3, kt = fid >> 2;
    int k = kt * 32 + (lane >> 4) * 8 + (j & 7) + (j >> 3) * 16;
    int n = nt * 16 + (lane & 15);
    f2[t] = f2bf(W2[k * 64 + n]);
  }
}

// ---------------- fused SAGE layer: WMMA GEMMs + bias + LN + ReLU ----------------
// block = 256 threads = 8 waves; 16 nodes per block; wave w owns output cols [16w,16w+16)
__global__ void __launch_bounds__(256)
k_sage(const float* __restrict__ agg, const float* __restrict__ invdeg,
       const unsigned short* __restrict__ hbf_in,
       const unsigned short* __restrict__ wfrag,
       const float* __restrict__ bl, const float* __restrict__ lng,
       const float* __restrict__ lnb, float* __restrict__ hout,
       unsigned short* __restrict__ hbf_out) {
  __shared__ __align__(16) unsigned short sMean[16][136];
  __shared__ __align__(16) unsigned short sH[16][136];
  __shared__ float sOut[16][129];
  __shared__ float sMu[16], sRs[16];

  const int tid = threadIdx.x;
  const int row0 = blockIdx.x * 16;

  // async-stage the 16 bf16 h-rows (4KB contiguous) straight into LDS
  {
    const unsigned short* gsrc = hbf_in + (size_t)row0 * HIDDEN;
    int r = tid >> 4, q = tid & 15;
    copy16_g2lds(gsrc + (size_t)tid * 8, &sH[r][q * 8]);
  }
  // mean = agg * inv_deg needs fp32 math -> synchronous convert
  for (int i = tid; i < 16 * HIDDEN; i += 256) {
    int r = i >> 7, c = i & 127;
    size_t g = (size_t)(row0 + r) * HIDDEN + c;
    sMean[r][c] = f2bf(agg[g] * invdeg[row0 + r]);
  }
  async_fence();
  __syncthreads();

  const int lane = tid & 31, wave = tid >> 5;
  const int m = lane & 15, half = lane >> 4;
  const int nt = wave;

  v8f acc = {0.f, 0.f, 0.f, 0.f, 0.f, 0.f, 0.f, 0.f};
#pragma unroll
  for (int kt = 0; kt < 4; ++kt) {
    ABFrag aM, aH, bL, bR;
    int ko = kt * 32 + half * 8;  // A layout per ISA: lane<16 -> K {0..7,16..23}
    aM.q[0] = *(const uint4*)&sMean[m][ko];
    aM.q[1] = *(const uint4*)&sMean[m][ko + 16];
    aH.q[0] = *(const uint4*)&sH[m][ko];
    aH.q[1] = *(const uint4*)&sH[m][ko + 16];
    const uint4* pl = (const uint4*)(wfrag + ((size_t)((0 * 4 + kt) * 8 + nt)) * 512 + lane * 16);
    const uint4* pr = (const uint4*)(wfrag + ((size_t)((4 + kt) * 8 + nt)) * 512 + lane * 16);
    bL.q[0] = pl[0]; bL.q[1] = pl[1];
    bR.q[0] = pr[0]; bR.q[1] = pr[1];
    if (kt < 3)
      __builtin_prefetch(wfrag + ((size_t)((kt + 1) * 8 + nt)) * 512 + lane * 16, 0, 1);
    acc = __builtin_amdgcn_wmma_f32_16x16x32_bf16(false, aM.v, false, bL.v, (short)0, acc, false, false);
    acc = __builtin_amdgcn_wmma_f32_16x16x32_bf16(false, aH.v, false, bR.v, (short)0, acc, false, false);
  }

  const int col = nt * 16 + m;
  const float bc = bl[col];
#pragma unroll
  for (int j = 0; j < 8; ++j) sOut[half * 8 + j][col] = acc[j] + bc;
  __syncthreads();

  if (tid < 16) {
    float mu = 0.f;
    for (int c = 0; c < HIDDEN; ++c) mu += sOut[tid][c];
    mu *= (1.0f / HIDDEN);
    float var = 0.f;
    for (int c = 0; c < HIDDEN; ++c) { float d = sOut[tid][c] - mu; var += d * d; }
    var *= (1.0f / HIDDEN);
    sMu[tid] = mu;
    sRs[tid] = rsqrtf(var + LN_EPS);
  }
  __syncthreads();

  for (int i = tid; i < 16 * HIDDEN; i += 256) {
    int r = i >> 7, c = i & 127;
    float v = (sOut[r][c] - sMu[r]) * sRs[r] * lng[c] + lnb[c];
    v = fmaxf(v, 0.f);
    size_t g = (size_t)(row0 + r) * HIDDEN + c;
    hout[g] = v;
    hbf_out[g] = f2bf(v);
  }
}

// ---------------- edge MLP: concat -> GEMM1(288x128) -> GEMM2(128x64) -> dot -> softplus ----
__global__ void __launch_bounds__(256)
k_edge_mlp(const unsigned short* __restrict__ hbf, const int* __restrict__ ei,
           const float* __restrict__ eattr,
           const unsigned short* __restrict__ w1f, const float* __restrict__ b1,
           const unsigned short* __restrict__ w2f, const float* __restrict__ b2,
           const float* __restrict__ W3, const float* __restrict__ b3,
           float* __restrict__ out) {
  __shared__ __align__(16) unsigned short sE[16][296];
  __shared__ __align__(16) unsigned short sZ1[16][136];
  __shared__ float sZ2[16][65];

  const int tid = threadIdx.x;
  const int e0 = blockIdx.x * 16;

  // async-gather bf16 rows: 16 edges x {src,dst} x 16 chunks of 16B = 512 copies
  for (int i = tid; i < 512; i += 256) {
    int r = i >> 5, sub = i & 31;
    int which = sub >> 4, q = sub & 15;
    int e = e0 + r;
    int node = which ? ei[N_EDGES + e] : ei[e];
    copy16_g2lds(hbf + (size_t)node * HIDDEN + q * 8, &sE[r][which * 128 + q * 8]);
  }
  // edge attributes + zero pad (needs fp32->bf16 convert, stay synchronous)
  for (int i = tid; i < 16 * 32; i += 256) {
    int r = i >> 5, c = i & 31;
    float v = (c < EDGE_F) ? eattr[(size_t)(e0 + r) * EDGE_F + c] : 0.f;
    sE[r][256 + c] = f2bf(v);
  }
  async_fence();
  __syncthreads();

  const int lane = tid & 31, wave = tid >> 5;
  const int m = lane & 15, half = lane >> 4;

  // GEMM1: [16,288] x [288,128], wave owns 16 columns
  {
    const int nt = wave;
    v8f acc = {0.f, 0.f, 0.f, 0.f, 0.f, 0.f, 0.f, 0.f};
#pragma unroll
    for (int kt = 0; kt < 9; ++kt) {
      ABFrag a, b;
      int ko = kt * 32 + half * 8;
      a.q[0] = *(const uint4*)&sE[m][ko];
      a.q[1] = *(const uint4*)&sE[m][ko + 16];
      const uint4* pb = (const uint4*)(w1f + ((size_t)(kt * 8 + nt)) * 512 + lane * 16);
      b.q[0] = pb[0]; b.q[1] = pb[1];
      if (kt < 8)
        __builtin_prefetch(w1f + ((size_t)((kt + 1) * 8 + nt)) * 512 + lane * 16, 0, 1);
      acc = __builtin_amdgcn_wmma_f32_16x16x32_bf16(false, a.v, false, b.v, (short)0, acc, false, false);
    }
    const int col = nt * 16 + m;
    const float bb = b1[col];
#pragma unroll
    for (int j = 0; j < 8; ++j)
      sZ1[half * 8 + j][col] = f2bf(fmaxf(acc[j] + bb, 0.f));
  }
  __syncthreads();

  // GEMM2: [16,128] x [128,64], waves 0..3
  if (wave < 4) {
    const int nt = wave;
    v8f acc = {0.f, 0.f, 0.f, 0.f, 0.f, 0.f, 0.f, 0.f};
#pragma unroll
    for (int kt = 0; kt < 4; ++kt) {
      ABFrag a, b;
      int ko = kt * 32 + half * 8;
      a.q[0] = *(const uint4*)&sZ1[m][ko];
      a.q[1] = *(const uint4*)&sZ1[m][ko + 16];
      const uint4* pb = (const uint4*)(w2f + ((size_t)(kt * 4 + nt)) * 512 + lane * 16);
      b.q[0] = pb[0]; b.q[1] = pb[1];
      acc = __builtin_amdgcn_wmma_f32_16x16x32_bf16(false, a.v, false, b.v, (short)0, acc, false, false);
    }
    const int col = nt * 16 + m;
    const float bb = b2[col];
#pragma unroll
    for (int j = 0; j < 8; ++j)
      sZ2[half * 8 + j][col] = fmaxf(acc[j] + bb, 0.f);
  }
  __syncthreads();

  // GEMM3 (64 -> 1) + stable softplus
  if (tid < 16) {
    float p = b3[0];
#pragma unroll 8
    for (int c = 0; c < 64; ++c) p += sZ2[tid][c] * W3[c];
    out[e0 + tid] = fmaxf(p, 0.f) + log1pf(expf(-fabsf(p)));
  }
}

// ---------------- host-side launcher ----------------
extern "C" void kernel_launch(void* const* d_in, const int* in_sizes, int n_in,
                              void* d_out, int out_size, void* d_ws, size_t ws_size,
                              hipStream_t stream) {
  (void)in_sizes; (void)n_in; (void)out_size; (void)ws_size;
  const float* x      = (const float*)d_in[0];
  const int*   ei     = (const int*)d_in[1];
  const float* eattr  = (const float*)d_in[2];
  const float* nodeW  = (const float*)d_in[3];
  const float* nodeB  = (const float*)d_in[4];
  const float* sageWl = (const float*)d_in[5];
  const float* sageBl = (const float*)d_in[6];
  const float* sageWr = (const float*)d_in[7];
  const float* lnG    = (const float*)d_in[8];
  const float* lnB    = (const float*)d_in[9];
  const float* W1     = (const float*)d_in[10];
  const float* b1     = (const float*)d_in[11];
  const float* W2     = (const float*)d_in[12];
  const float* b2     = (const float*)d_in[13];
  const float* W3     = (const float*)d_in[14];
  const float* b3     = (const float*)d_in[15];
  float* out = (float*)d_out;

  char* ws = (char*)d_ws;
  size_t off = 0;
  auto walloc = [&](size_t bytes) -> char* {
    char* p = ws + off;
    off = (off + bytes + 255) & ~(size_t)255;
    return p;
  };
  float* invdeg = (float*)walloc((size_t)N_NODES * 4);
  float* hA     = (float*)walloc((size_t)N_NODES * HIDDEN * 4);
  float* hB     = (float*)walloc((size_t)N_NODES * HIDDEN * 4);
  unsigned short* hbfA = (unsigned short*)walloc((size_t)N_NODES * HIDDEN * 2);
  unsigned short* hbfB = (unsigned short*)walloc((size_t)N_NODES * HIDDEN * 2);
  float* agg    = (float*)walloc((size_t)N_NODES * HIDDEN * 4);
  unsigned short* fsage = (unsigned short*)walloc((size_t)SAGE_FRAG_ELEMS * 2);
  unsigned short* f1    = (unsigned short*)walloc((size_t)W1_FRAG_ELEMS * 2);
  unsigned short* f2    = (unsigned short*)walloc((size_t)W2_FRAG_ELEMS * 2);

  // degrees
  k_zero4<<<(N_NODES / 4 + 255) / 256, 256, 0, stream>>>((float4*)invdeg, N_NODES / 4);
  k_degree<<<(N_EDGES + 255) / 256, 256, 0, stream>>>(ei, invdeg);
  k_invdeg<<<(N_NODES + 255) / 256, 256, 0, stream>>>(invdeg);

  // weight prep (bf16 WMMA B fragments)
  {
    int total = SAGE_FRAG_ELEMS + W1_FRAG_ELEMS + W2_FRAG_ELEMS;  // 143360
    k_prep_weights<<<(total + 255) / 256, 256, 0, stream>>>(sageWl, sageWr, W1, W2, fsage, f1, f2);
  }

  // node projection (fp32 + bf16 shadow)
  k_node_proj<<<(N_NODES * HIDDEN + 255) / 256, 256, 0, stream>>>(x, nodeW, nodeB, hA, hbfA);

  float* cur = hA;            float* nxt = hB;
  unsigned short* curbf = hbfA; unsigned short* nxtbf = hbfB;
  for (int layer = 0; layer < N_LAYERS; ++layer) {
    k_zero4<<<(N_NODES * HIDDEN / 4 + 255) / 256, 256, 0, stream>>>((float4*)agg, N_NODES * HIDDEN / 4);
    k_scatter<<<(N_EDGES * 32) / 256, 256, 0, stream>>>(cur, ei, agg);
    k_sage<<<N_NODES / 16, 256, 0, stream>>>(agg, invdeg, curbf,
        fsage + (size_t)layer * 2 * 4 * 8 * 512,
        sageBl + layer * HIDDEN, lnG + layer * HIDDEN, lnB + layer * HIDDEN,
        nxt, nxtbf);
    { float* t = cur; cur = nxt; nxt = t; }
    { unsigned short* t = curbf; curbf = nxtbf; nxtbf = t; }
  }

  k_edge_mlp<<<N_EDGES / 16, 256, 0, stream>>>(curbf, ei, eattr, f1, b1, f2, b2, W3, b3, out);
}
